// GPTLanguageModel_17205638988576
// MI455X (gfx1250) — matrix operand order repaired and checked by reading, not compiled, not run on hardware
//
#include <hip/hip_runtime.h>

// ---------------- model dims ----------------
#define E_    384
#define T_    256
#define B_    16
#define H_    6
#define HS_   64
#define L_    6
#define V_    50257
#define VPAD  50304          // 393 * 128
#define M_    (B_*T_)        // 4096 rows
#define FF    1536

typedef __attribute__((ext_vector_type(16))) __bf16        bfx16;
typedef __attribute__((ext_vector_type(8)))  float         fx8;
typedef __attribute__((ext_vector_type(4)))  unsigned int  ux4;
typedef __attribute__((ext_vector_type(4)))  unsigned int  v4u;
typedef __attribute__((ext_vector_type(8)))  int           v8i_;
typedef __attribute__((ext_vector_type(4)))  int           v4i_;

__device__ __forceinline__ unsigned short f2bf(float f) {
  unsigned u = __float_as_uint(f);
  u += 0x7FFFu + ((u >> 16) & 1u);               // round-to-nearest-even
  return (unsigned short)(u >> 16);
}

// ---- TDM: issue async 2D tile load (tile_dim0=64 elems along K, tile_dim1=128 rows)
// from bf16 weight matrix Wt[Npad][K] into LDS at byte offset ldsaddr.
__device__ __forceinline__ void tdm_issue_b(const unsigned short* Wt, int n0b, int k0,
                                            unsigned K, unsigned Npad, unsigned ldsaddr) {
  unsigned long long ga = (unsigned long long)(uintptr_t)(Wt + (long)n0b * K + k0);
  // D# group0: count=1 | lds_addr | global_addr[56:0] | type=2
  v4u g0 = { 1u,
             ldsaddr,
             (unsigned)(ga & 0xFFFFFFFFull),
             (unsigned)((ga >> 32) & 0x01FFFFFFull) | 0x80000000u };
  // D# group1: data_size=1(2B); tensor_dim0=K; tensor_dim1=Npad; tile_dim0=64;
  //            tile_dim1=128; tensor_dim0_stride=K (elements)
  v8i_ g1 = { (int)0x00010000u,
              (int)((K & 0xFFFFu) << 16),
              (int)(((Npad & 0xFFFFu) << 16) | ((K >> 16) & 0xFFFFu)),
              (int)((64u << 16) | ((Npad >> 16) & 0xFFFFu)),
              (int)128u,
              (int)K,
              0, 0 };
  v4i_ z4 = { 0, 0, 0, 0 };
#if __clang_major__ >= 23
  v8i_ z8 = { 0, 0, 0, 0, 0, 0, 0, 0 };
  __builtin_amdgcn_tensor_load_to_lds(g0, g1, z4, z4, z8, 0);
#else
  __builtin_amdgcn_tensor_load_to_lds(g0, g1, z4, z4, 0);
#endif
}

// ---------------- embedding: x = tok_emb[idx] + pos_emb[:T] ----------------
__global__ void k_embed(const int* __restrict__ idx, const float* __restrict__ tok,
                        const float* __restrict__ pos, float* __restrict__ x) {
  int i = blockIdx.x * 256 + threadIdx.x;        // over M_*E_
  if (i >= M_*E_) return;
  int row = i / E_, e = i - row*E_;
  int t = row % T_;
  x[i] = tok[(long)idx[row]*E_ + e] + pos[(long)t*E_ + e];
}

// ---- weight convert + transpose: W[K][N] f32 -> Wt[Npad][K] bf16 (zero pad) ----
__global__ void k_wt(const float* __restrict__ W, unsigned short* __restrict__ Wt,
                     int K, int N, int Npad) {
  long i = (long)blockIdx.x * 256 + threadIdx.x; // over Npad*K
  if (i >= (long)Npad * K) return;
  int n = (int)(i / K), k = (int)(i - (long)n * K);
  float v = (n < N) ? W[(long)k * N + n] : 0.f;
  Wt[i] = f2bf(v);
}

// ---------------- layernorm (fp32 in, bf16 out) ----------------
__global__ void k_ln(const float* __restrict__ x, const float* __restrict__ g,
                     const float* __restrict__ b, unsigned short* __restrict__ out) {
  int row = blockIdx.x, t = threadIdx.x;         // 128 threads, E_=384 -> 3 each
  __shared__ float s1[128], s2[128];
  const float* p = x + (long)row * E_;
  float v0 = p[t], v1 = p[t + 128], v2 = p[t + 256];
  s1[t] = v0 + v1 + v2;
  s2[t] = v0*v0 + v1*v1 + v2*v2;
  __syncthreads();
  for (int o = 64; o > 0; o >>= 1) {
    if (t < o) { s1[t] += s1[t + o]; s2[t] += s2[t + o]; }
    __syncthreads();
  }
  float mu  = s1[0] * (1.f / E_);
  float var = s2[0] * (1.f / E_) - mu * mu;
  float r   = rsqrtf(var + 1e-5f);
  unsigned short* op = out + (long)row * E_;
  op[t]       = f2bf((v0 - mu) * r * g[t]       + b[t]);
  op[t + 128] = f2bf((v1 - mu) * r * g[t + 128] + b[t + 128]);
  op[t + 256] = f2bf((v2 - mu) * r * g[t + 256] + b[t + 256]);
}

// ---------------- WMMA bf16 GEMM with TDM double-buffered B tiles ----------------
// C[M][N] = A[M][K] (bf16) @ Wt[N][K]^T (bf16) + bias, optional relu/residual/NT store.
// Block: 256 thr = 8 waves; wave tile 16x64; block tile 64x128; grid=(M/64, Npad/128).
// Wave 0 drives the Tensor Data Mover: 128x64-elem B tile per K-step into LDS
// (double buffered, TENSORcnt); all waves consume via ds_load_b128 after the barrier.
// B fragments are preloaded into registers before the WMMA burst so the matrix pipe
// runs back-to-back instead of waiting on DScnt per instruction.
__global__ __launch_bounds__(256)
void k_gemm(const unsigned short* __restrict__ A, const unsigned short* __restrict__ Wt,
            const float* __restrict__ bias, const float* __restrict__ resid,
            float* __restrict__ outF, unsigned short* __restrict__ outBf,
            int N, int K, int Npad, int ldOut, int relu, int ntstore)
{
  __shared__ unsigned short bsm[2][128 * 64];    // 2 x 16 KB B tiles
  int lane = threadIdx.x & 31;
  int wid  = threadIdx.x >> 5;                   // 0..7
  int hi   = lane >> 4;                          // 0/1 (lane half)
  int lr   = lane & 15;
  int m0  = blockIdx.x * 64 + (wid & 3) * 16;
  int nw  = (wid >> 2) * 64;                     // wave N offset within block tile
  int n0b = blockIdx.y * 128;

  unsigned lds0 = (unsigned)(uintptr_t)&bsm[0][0];
  unsigned lds1 = (unsigned)(uintptr_t)&bsm[1][0];

  const unsigned short* arow = A + (long)(m0 + lr) * K;

  if (wid == 0) tdm_issue_b(Wt, n0b, 0, (unsigned)K, (unsigned)Npad, lds0);

  fx8 acc[4] = {};
  int p = 0;
  for (int k0 = 0; k0 < K; k0 += 64) {
    if (wid == 0) {
      if (k0 + 64 < K) {
        tdm_issue_b(Wt, n0b, k0 + 64, (unsigned)K, (unsigned)Npad, p ? lds0 : lds1);
        __builtin_amdgcn_s_wait_tensorcnt(1);    // current tile complete
      } else {
        __builtin_amdgcn_s_wait_tensorcnt(0);
      }
    }
    __syncthreads();                             // tile p visible to all waves

    if (k0 + 64 < K) __builtin_prefetch(arow + k0 + 64, 0, 1);   // A stream prefetch
    const unsigned short* bs = &bsm[p][0];       // LDS tile: [n_local][64]

#pragma unroll
    for (int kk = 0; kk < 64; kk += 32) {
      // A fragment (16x32 bf16): lanes 0-15: K {+0..+7, +16..+23};
      // lanes 16-31: K {+8..+15, +24..+31} (ISA 7.12.2).
      union { bfx16 v; ux4 u[2]; } a;
      a.u[0] = *(const ux4*)(arow + k0 + kk + hi * 8);
      a.u[1] = *(const ux4*)(arow + k0 + kk + 16 + hi * 8);

      // Preload all 4 B fragments (lane n = column n; lanes 0-15 K {+0..+7, +16..+23},
      // lanes 16-31 K {+8..+15, +24..+31}, matching A).
      union { bfx16 v; ux4 u[2]; } bfr[4];
#pragma unroll
      for (int j = 0; j < 4; ++j) {
        const unsigned short* brow = bs + (nw + j * 16 + lr) * 64 + kk + hi * 8;
        bfr[j].u[0] = *(const ux4*)(brow);
        bfr[j].u[1] = *(const ux4*)(brow + 16);
      }
      // WMMA burst, A reused 4x
#pragma unroll
      for (int j = 0; j < 4; ++j)
        acc[j] = __builtin_amdgcn_wmma_f32_16x16x32_bf16(
                   false, a.v, false, bfr[j].v, (short)0, acc[j], false, false);
    }
    __syncthreads();                             // all reads of tile p done before reuse
    p ^= 1;
  }

  // C/D layout: VGPR r, lane l -> row = r + 8*(l>>4), col = l&15.
#pragma unroll
  for (int j = 0; j < 4; ++j) {
    int col = n0b + nw + j * 16 + lr;
    if (col >= N) continue;
    float bb = bias ? bias[col] : 0.f;
#pragma unroll
    for (int r = 0; r < 8; ++r) {
      int rowi = m0 + r + 8 * hi;
      float v = acc[j][r] + bb;
      if (relu) v = fmaxf(v, 0.f);
      long o = (long)rowi * ldOut + col;
      if (resid) v += resid[o];
      if (outF) {
        if (ntstore) __builtin_nontemporal_store(v, &outF[o]);   // logits stream
        else         outF[o] = v;
      }
      if (outBf) outBf[o] = f2bf(v);
    }
  }
}

// ---------------- causal attention, online softmax, 1 thread / (head,row) ----------------
__global__ void k_attn(const float* __restrict__ q, const float* __restrict__ k,
                       const float* __restrict__ v, unsigned short* __restrict__ out) {
  int id = blockIdx.x * 256 + threadIdx.x;       // M_*H_ = 24576
  if (id >= M_ * H_) return;
  int h   = id / M_;                             // consecutive lanes: consecutive rows, same head
  int row = id - h * M_;
  int b = row / T_, t = row - b * T_;
  const float sc = 0.051031036307982884f;        // E^-0.5 (as in reference)

  const float* qp = q + (long)row * E_ + h * HS_;
  float qr[HS_], acc[HS_];
#pragma unroll
  for (int d = 0; d < HS_; ++d) { qr[d] = qp[d]; acc[d] = 0.f; }
  float mx = -3.0e38f, s = 0.f;
  long base = (long)b * T_ * E_ + h * HS_;
  for (int j = 0; j <= t; ++j) {
    const float* kp = k + base + (long)j * E_;
    float dot = 0.f;
#pragma unroll
    for (int d = 0; d < HS_; ++d) dot += qr[d] * kp[d];
    float w    = dot * sc;
    float nm   = fmaxf(mx, w);
    float corr = __expf(mx - nm);
    float pw   = __expf(w - nm);
    s = s * corr + pw;
    const float* vp = v + base + (long)j * E_;
#pragma unroll
    for (int d = 0; d < HS_; ++d) acc[d] = acc[d] * corr + pw * vp[d];
    mx = nm;
  }
  float inv = 1.f / s;
  unsigned short* op = out + (long)row * E_ + h * HS_;
#pragma unroll
  for (int d = 0; d < HS_; ++d) op[d] = f2bf(acc[d] * inv);
}

// ---------------- per-row -logp[target] over V logits ----------------
__global__ void k_lossrow(const float* __restrict__ logits, const int* __restrict__ tgt,
                          float* __restrict__ rl) {
  int row = blockIdx.x, t = threadIdx.x;         // 256 threads
  const float* p = logits + (long)row * V_;
  __shared__ float sm[256];
  float mx = -3.0e38f;
  for (int i = t; i < V_; i += 256) mx = fmaxf(mx, p[i]);
  sm[t] = mx; __syncthreads();
  for (int o = 128; o > 0; o >>= 1) { if (t < o) sm[t] = fmaxf(sm[t], sm[t + o]); __syncthreads(); }
  mx = sm[0]; __syncthreads();
  float s = 0.f;
  for (int i = t; i < V_; i += 256) s += __expf(p[i] - mx);
  sm[t] = s; __syncthreads();
  for (int o = 128; o > 0; o >>= 1) { if (t < o) sm[t] += sm[t + o]; __syncthreads(); }
  if (t == 0) rl[row] = mx + __logf(sm[0]) - p[tgt[row]];
}

__global__ void k_lossred(const float* __restrict__ rl, float* __restrict__ loss) {
  int t = threadIdx.x;
  __shared__ float sm[256];
  float s = 0.f;
  for (int i = t; i < M_; i += 256) s += rl[i];
  sm[t] = s; __syncthreads();
  for (int o = 128; o > 0; o >>= 1) { if (t < o) sm[t] += sm[t + o]; __syncthreads(); }
  if (t == 0) *loss = sm[0] / (float)M_;
}

// ---------------- launcher ----------------
extern "C" void kernel_launch(void* const* d_in, const int* in_sizes, int n_in,
                              void* d_out, int out_size, void* d_ws, size_t ws_size,
                              hipStream_t stream) {
  const int*   idx   = (const int*)  d_in[0];
  const int*   tgt   = (const int*)  d_in[1];
  const float* tok   = (const float*)d_in[2];
  const float* pos   = (const float*)d_in[3];
  const float* Wq    = (const float*)d_in[4];
  const float* Wk    = (const float*)d_in[5];
  const float* Wv    = (const float*)d_in[6];
  const float* Wp    = (const float*)d_in[7];
  const float* bproj = (const float*)d_in[8];
  const float* W1    = (const float*)d_in[9];
  const float* b1    = (const float*)d_in[10];
  const float* W2    = (const float*)d_in[11];
  const float* b2    = (const float*)d_in[12];
  const float* ln1g  = (const float*)d_in[13];
  const float* ln1b  = (const float*)d_in[14];
  const float* ln2g  = (const float*)d_in[15];
  const float* ln2b  = (const float*)d_in[16];
  const float* lnfg  = (const float*)d_in[17];
  const float* lnfb  = (const float*)d_in[18];
  const float* Wh    = (const float*)d_in[19];
  const float* bh    = (const float*)d_in[20];
  float* logits = (float*)d_out;
  float* lossp  = (float*)d_out + (long)M_ * V_;

  char* ws = (char*)d_ws;
  size_t off = 0;
  auto nxt = [&](size_t bytes) { size_t r = off; off += (bytes + 255) & ~(size_t)255; return r; };
  float*          xf   = (float*)         (ws + nxt((size_t)M_*E_*4));
  unsigned short* hbf  = (unsigned short*)(ws + nxt((size_t)M_*E_*2));
  float*          qf   = (float*)         (ws + nxt((size_t)M_*E_*4));
  float*          kf   = (float*)         (ws + nxt((size_t)M_*E_*4));
  float*          vf   = (float*)         (ws + nxt((size_t)M_*E_*4));
  unsigned short* abf  = (unsigned short*)(ws + nxt((size_t)M_*E_*2));
  unsigned short* f1bf = (unsigned short*)(ws + nxt((size_t)M_*FF*2));
  unsigned short* wqt  = (unsigned short*)(ws + nxt((size_t)E_*E_*2));
  unsigned short* wkt  = (unsigned short*)(ws + nxt((size_t)E_*E_*2));
  unsigned short* wvt  = (unsigned short*)(ws + nxt((size_t)E_*E_*2));
  unsigned short* wpt  = (unsigned short*)(ws + nxt((size_t)E_*E_*2));
  unsigned short* w1t  = (unsigned short*)(ws + nxt((size_t)FF*E_*2));
  unsigned short* w2t  = (unsigned short*)(ws + nxt((size_t)E_*FF*2));
  unsigned short* wht  = (unsigned short*)(ws + nxt((size_t)VPAD*E_*2));
  float*          rls  = (float*)         (ws + nxt((size_t)M_*4));

  dim3 blk(256);
  k_embed<<<(M_*E_ + 255)/256, blk, 0, stream>>>(idx, tok, pos, xf);
  k_wt<<<(int)(((long)VPAD*E_ + 255)/256), blk, 0, stream>>>(Wh, wht, E_, V_, VPAD);

  dim3 g1(M_/64, E_/128);    // 384-wide outputs
  dim3 g2(M_/64, FF/128);    // 1536-wide outputs
  for (int l = 0; l < L_; ++l) {
    const float* Wq_l = Wq + (long)l*E_*E_;
    const float* Wk_l = Wk + (long)l*E_*E_;
    const float* Wv_l = Wv + (long)l*E_*E_;
    const float* Wp_l = Wp + (long)l*E_*E_;
    const float* W1_l = W1 + (long)l*E_*FF;
    const float* W2_l = W2 + (long)l*FF*E_;
    k_wt<<<(E_*E_ + 255)/256, blk, 0, stream>>>(Wq_l, wqt, E_, E_, E_);
    k_wt<<<(E_*E_ + 255)/256, blk, 0, stream>>>(Wk_l, wkt, E_, E_, E_);
    k_wt<<<(E_*E_ + 255)/256, blk, 0, stream>>>(Wv_l, wvt, E_, E_, E_);
    k_wt<<<(E_*E_ + 255)/256, blk, 0, stream>>>(Wp_l, wpt, E_, E_, E_);
    k_wt<<<(E_*FF + 255)/256, blk, 0, stream>>>(W1_l, w1t, E_, FF, FF);
    k_wt<<<(FF*E_ + 255)/256, blk, 0, stream>>>(W2_l, w2t, FF, E_, E_);

    k_ln<<<M_, 128, 0, stream>>>(xf, ln1g + l*E_, ln1b + l*E_, hbf);
    k_gemm<<<g1, blk, 0, stream>>>(hbf, wqt, nullptr, nullptr, qf, nullptr, E_, E_, E_, E_, 0, 0);
    k_gemm<<<g1, blk, 0, stream>>>(hbf, wkt, nullptr, nullptr, kf, nullptr, E_, E_, E_, E_, 0, 0);
    k_gemm<<<g1, blk, 0, stream>>>(hbf, wvt, nullptr, nullptr, vf, nullptr, E_, E_, E_, E_, 0, 0);
    k_attn<<<(M_*H_)/256, blk, 0, stream>>>(qf, kf, vf, abf);
    k_gemm<<<g1, blk, 0, stream>>>(abf, wpt, bproj + l*E_, xf, xf, nullptr, E_, E_, E_, E_, 0, 0);
    k_ln<<<M_, 128, 0, stream>>>(xf, ln2g + l*E_, ln2b + l*E_, hbf);
    k_gemm<<<g2, blk, 0, stream>>>(hbf, w1t, b1 + l*FF, nullptr, nullptr, f1bf, FF, E_, FF, FF, 1, 0);
    k_gemm<<<g1, blk, 0, stream>>>(f1bf, w2t, b2 + l*E_, xf, xf, nullptr, E_, FF, E_, E_, 0, 0);
  }

  k_ln<<<M_, 128, 0, stream>>>(xf, lnfg, lnfb, hbf);
  dim3 gh(M_/64, VPAD/128);
  k_gemm<<<gh, blk, 0, stream>>>(hbf, wht, bh, nullptr, logits, nullptr, V_, E_, VPAD, V_, 0, 1);

  k_lossrow<<<M_, blk, 0, stream>>>(logits, tgt, rls);
  k_lossred<<<1, blk, 0, stream>>>(rls, lossp);
}
